// VQA_model_52020643889336
// MI455X (gfx1250) — matrix operand order repaired and checked by reading, not compiled, not run on hardware
//
#include <hip/hip_runtime.h>
#include <hip/hip_bf16.h>

// ---------------------------------------------------------------------------
// Types for CDNA5 WMMA (wave32)
// ---------------------------------------------------------------------------
typedef __attribute__((ext_vector_type(16))) __bf16 v16bf;
typedef __attribute__((ext_vector_type(8)))  float  v8f;

union Frag16 { v16bf v; unsigned int u[8]; };

__device__ __forceinline__ unsigned int f2bf_u(float f) {
  unsigned int u = __float_as_uint(f);
  return (u + 0x7FFFu + ((u >> 16) & 1u)) >> 16;   // round-nearest-even
}
__device__ __forceinline__ float leakyf(float x) { return x >= 0.f ? x : 0.01f * x; }
__device__ __forceinline__ float sigf(float x)   { return 1.f / (1.f + __expf(-x)); }

// ---------------------------------------------------------------------------
// Generic tiled GEMM:  C = epilogue( A(M,K) @ W(N,K)^T )
//   bf16 compute via v_wmma_f32_16x16x32_bf16, f32 accumulate.
//   mode 0: C = X (+bias)(+bias2)(+addend)
//   mode 1: C = leaky(X + bias ...)
//   mode 2: C[m] += sum_n leaky(X+bias) * vec[n]   (fused rank-1 reduction)
// Block: 256 threads = 8 waves, 128x128 tile; wave = 32x64 sub-tile.
// ---------------------------------------------------------------------------
#define TM 128
#define TN 128
#define TK 32
#define LDT 40   // padded LDS row stride (bf16 elems): 80B, 16B aligned

__global__ __launch_bounds__(256) void gemm_wmma_kernel(
    const float* __restrict__ A, int lda,
    const float* __restrict__ W,             // (N,K) row-major
    const float* __restrict__ bias,          // len N or null
    const float* __restrict__ bias2,         // len N or null
    const float* __restrict__ addend, int ldadd,  // (M,N) or null
    const float* __restrict__ vec,           // len N (mode 2)
    float* __restrict__ C, int ldc,          // mode 2: C is out[M]
    int M, int N, int K, int mode)
{
  __shared__ unsigned short As[TM * LDT];
  __shared__ unsigned short Bs[TN * LDT];
  __shared__ float red[TM];

  const int tid  = threadIdx.x;
  const int lane = tid & 31, wv = tid >> 5;
  const int wr = wv & 3, wc = wv >> 2;          // 4x2 wave grid
  const int hi = lane >> 4, lr = lane & 15;
  const int m0 = blockIdx.y * TM;
  const int n0 = blockIdx.x * TN;

  v8f acc[2][4] = {};
  if (mode == 2 && tid < TM) red[tid] = 0.f;

  // staging decomposition: 4 consecutive k per thread, 4 iterations cover a tile
  const int sr = tid >> 3;          // 0..31  (row group base; +32 per iter)
  const int sk = (tid & 7) * 4;     // 0,4,...,28

  for (int k0 = 0; k0 < K; k0 += TK) {
    const bool fullK = (k0 + TK <= K);
    const bool fullA = fullK && (m0 + TM <= M);
    const bool fullB = fullK && (n0 + TN <= N);

    // ---- stage A tile (f32 -> bf16, packed b64 stores) ----
    #pragma unroll
    for (int it = 0; it < 4; ++it) {
      int r = sr + it * 32;
      float v0, v1, v2, v3;
      if (fullA) {
        const float* p = &A[(long)(m0 + r) * lda + k0 + sk];
        v0 = p[0]; v1 = p[1]; v2 = p[2]; v3 = p[3];
      } else {
        int gm = m0 + r, gk = k0 + sk;
        const float* p = &A[(long)gm * lda + gk];
        bool mok = gm < M;
        v0 = (mok && gk + 0 < K) ? p[0] : 0.f;
        v1 = (mok && gk + 1 < K) ? p[1] : 0.f;
        v2 = (mok && gk + 2 < K) ? p[2] : 0.f;
        v3 = (mok && gk + 3 < K) ? p[3] : 0.f;
      }
      unsigned int p0 = f2bf_u(v0) | (f2bf_u(v1) << 16);
      unsigned int p1 = f2bf_u(v2) | (f2bf_u(v3) << 16);
      unsigned long long pk = (unsigned long long)p0 | ((unsigned long long)p1 << 32);
      *(unsigned long long*)&As[r * LDT + sk] = pk;
    }
    // ---- stage W tile (as [n][k]) ----
    #pragma unroll
    for (int it = 0; it < 4; ++it) {
      int n = sr + it * 32;
      float v0, v1, v2, v3;
      if (fullB) {
        const float* p = &W[(long)(n0 + n) * K + k0 + sk];
        v0 = p[0]; v1 = p[1]; v2 = p[2]; v3 = p[3];
      } else {
        int gn = n0 + n, gk = k0 + sk;
        const float* p = &W[(long)gn * K + gk];
        bool nok = gn < N;
        v0 = (nok && gk + 0 < K) ? p[0] : 0.f;
        v1 = (nok && gk + 1 < K) ? p[1] : 0.f;
        v2 = (nok && gk + 2 < K) ? p[2] : 0.f;
        v3 = (nok && gk + 3 < K) ? p[3] : 0.f;
      }
      unsigned int p0 = f2bf_u(v0) | (f2bf_u(v1) << 16);
      unsigned int p1 = f2bf_u(v2) | (f2bf_u(v3) << 16);
      unsigned long long pk = (unsigned long long)p0 | ((unsigned long long)p1 << 32);
      *(unsigned long long*)&Bs[n * LDT + sk] = pk;
    }
    // prefetch next K-tile (global_prefetch_b8)
    if (k0 + TK < K) {
      int r  = tid >> 1;
      int kk = k0 + TK + (tid & 1) * 16;
      if (m0 + r < M && kk < K) __builtin_prefetch(&A[(long)(m0 + r) * lda + kk], 0, 1);
      if (n0 + r < N && kk < K) __builtin_prefetch(&W[(long)(n0 + r) * K   + kk], 0, 1);
    }
    __syncthreads();

    // ---- build fragments per ISA 7.12.2 bf16 layouts ----
    Frag16 af[2], bfr[4];
    #pragma unroll
    for (int mi = 0; mi < 2; ++mi) {
      const unsigned short* ar = &As[(wr * 32 + mi * 16 + lr) * LDT];
      #pragma unroll
      for (int g = 0; g < 2; ++g)
        #pragma unroll
        for (int p = 0; p < 4; ++p)
          af[mi].u[g * 4 + p] = *(const unsigned int*)&ar[g * 16 + hi * 8 + p * 2];
    }
    #pragma unroll
    for (int ni = 0; ni < 4; ++ni) {
      const unsigned short* br = &Bs[(wc * 64 + ni * 16 + lr) * LDT];
      #pragma unroll
      for (int p = 0; p < 8; ++p)
        bfr[ni].u[p] = *(const unsigned int*)&br[(p >> 2) * 16 + hi * 8 + (p & 3) * 2];
    }
    #pragma unroll
    for (int mi = 0; mi < 2; ++mi)
      #pragma unroll
      for (int ni = 0; ni < 4; ++ni)
        acc[mi][ni] = __builtin_amdgcn_wmma_f32_16x16x32_bf16(
            false, af[mi].v, false, bfr[ni].v, (short)0, acc[mi][ni], false, false);
    __syncthreads();
  }

  // ---- epilogue ----
  #pragma unroll
  for (int mi = 0; mi < 2; ++mi) {
    #pragma unroll
    for (int ni = 0; ni < 4; ++ni) {
      #pragma unroll
      for (int i = 0; i < 8; ++i) {
        int rl = wr * 32 + mi * 16 + hi * 8 + i;
        int cl = wc * 64 + ni * 16 + lr;
        int row = m0 + rl, col = n0 + cl;
        if (row >= M || col >= N) continue;
        float v = acc[mi][ni][i];
        if (bias)  v += bias[col];
        if (bias2) v += bias2[col];
        if (mode == 2) {
          atomicAdd(&red[rl], leakyf(v) * vec[col]);
        } else {
          if (addend) v += addend[(long)row * ldadd + col];
          if (mode == 1) v = leakyf(v);
          C[(long)row * ldc + col] = v;
        }
      }
    }
  }
  if (mode == 2) {
    __syncthreads();
    if (tid < TM && m0 + tid < M) atomicAdd(&C[m0 + tid], red[tid]);
  }
}

// ---------------------------------------------------------------------------
// Supporting kernels
// ---------------------------------------------------------------------------
__global__ void fill_kernel(float* out, long n, float val, const float* addp) {
  long i = (long)blockIdx.x * blockDim.x + threadIdx.x;
  if (i < n) out[i] = val + (addp ? addp[0] : 0.f);
}

__global__ void embed_kernel(const float* __restrict__ emb, const int* __restrict__ idx,
                             float* __restrict__ out, int E) {
  int m = blockIdx.x;
  long id = idx[m];
  for (int e = threadIdx.x; e < E; e += blockDim.x)
    out[(long)m * E + e] = emb[id * E + e];
}

// gates (128 x 2048) -> update c (128 x 512), write h into enc_q column block
__global__ void lstm_pointwise_kernel(const float* __restrict__ gates,
                                      float* __restrict__ c,
                                      float* __restrict__ encq, int t, int colofs) {
  int idx = blockIdx.x * blockDim.x + threadIdx.x;
  if (idx >= 128 * 512) return;
  int b = idx >> 9, j = idx & 511;
  const float* g = gates + (long)b * 2048;
  float ig = sigf(g[j]), fg = sigf(g[512 + j]);
  float gg = tanhf(g[1024 + j]), og = sigf(g[1536 + j]);
  float cn = fg * c[idx] + ig * gg;
  c[idx] = cn;
  encq[((long)b * 23 + t) * 1024 + colofs + j] = og * tanhf(cn);
}

// out[m] = scale * dot(nrm(P1[m]), nrm(P2[m])), rows of length 512
__global__ void rowdot_nrm_kernel(const float* __restrict__ P1, const float* __restrict__ P2,
                                  float* __restrict__ out, float scale) {
  int m = blockIdx.x, tid = threadIdx.x;
  __shared__ float r0[256], r1[256], r2[256];
  float s12 = 0, s11 = 0, s22 = 0;
  for (int j = tid; j < 512; j += 256) {
    float a = P1[(long)m * 512 + j], b = P2[(long)m * 512 + j];
    s12 += a * b; s11 += a * a; s22 += b * b;
  }
  r0[tid] = s12; r1[tid] = s11; r2[tid] = s22; __syncthreads();
  for (int o = 128; o > 0; o >>= 1) {
    if (tid < o) { r0[tid] += r0[tid + o]; r1[tid] += r1[tid + o]; r2[tid] += r2[tid + o]; }
    __syncthreads();
  }
  if (tid == 0) {
    float n1 = fmaxf(sqrtf(r1[0]), 1e-8f), n2 = fmaxf(sqrtf(r2[0]), 1e-8f);
    out[m] = scale * r0[0] / (n1 * n2);
  }
}

__global__ void row_norms_kernel(const float* __restrict__ P, float* __restrict__ out) {
  int m = blockIdx.x, tid = threadIdx.x;
  __shared__ float red[256];
  float s = 0;
  for (int j = tid; j < 512; j += 256) { float a = P[(long)m * 512 + j]; s += a * a; }
  red[tid] = s; __syncthreads();
  for (int o = 128; o > 0; o >>= 1) { if (tid < o) red[tid] += red[tid + o]; __syncthreads(); }
  if (tid == 0) out[m] = fmaxf(sqrtf(red[0]), 1e-8f);
}

// S[b,d] = sum_q P[(b*Q+q), d] / norms[b*Q+q]
__global__ void sum_nrm_kernel(const float* __restrict__ P, const float* __restrict__ norms,
                               float* __restrict__ S, int Q) {
  int b = blockIdx.x;
  int d = blockIdx.y * 256 + threadIdx.x;
  float s = 0;
  for (int q = 0; q < Q; ++q) s += P[((long)(b * Q + q)) * 512 + d] / norms[b * Q + q];
  S[(long)b * 512 + d] = s;
}

// out[m] = dot(nrm(P[m]), S[b]),  b = m / rowsPerB
__global__ void rowdot_vec_kernel(const float* __restrict__ P, const float* __restrict__ S,
                                  float* __restrict__ out, int rowsPerB) {
  int m = blockIdx.x, tid = threadIdx.x;
  int b = m / rowsPerB;
  __shared__ float r0[256], r1[256];
  float sd = 0, sn = 0;
  for (int j = tid; j < 512; j += 256) {
    float a = P[(long)m * 512 + j];
    sd += a * S[(long)b * 512 + j]; sn += a * a;
  }
  r0[tid] = sd; r1[tid] = sn; __syncthreads();
  for (int o = 128; o > 0; o >>= 1) {
    if (tid < o) { r0[tid] += r0[tid + o]; r1[tid] += r1[tid + o]; }
    __syncthreads();
  }
  if (tid == 0) out[m] = r0[0] / fmaxf(sqrtf(r1[0]), 1e-8f);
}

__global__ void softmax_rows_kernel(float* __restrict__ X, int N) {
  int b = blockIdx.x, tid = threadIdx.x;
  __shared__ float red[256];
  float mx = -3.4e38f;
  for (int j = tid; j < N; j += 256) mx = fmaxf(mx, X[(long)b * N + j]);
  red[tid] = mx; __syncthreads();
  for (int o = 128; o > 0; o >>= 1) { if (tid < o) red[tid] = fmaxf(red[tid], red[tid + o]); __syncthreads(); }
  float gmx = red[0]; __syncthreads();
  float s = 0;
  for (int j = tid; j < N; j += 256) s += __expf(X[(long)b * N + j] - gmx);
  red[tid] = s; __syncthreads();
  for (int o = 128; o > 0; o >>= 1) { if (tid < o) red[tid] += red[tid + o]; __syncthreads(); }
  float inv = 1.f / red[0];
  for (int j = tid; j < N; j += 256) X[(long)b * N + j] = __expf(X[(long)b * N + j] - gmx) * inv;
}

// out[b, off+d] = sum_r w[b,r] * X[(b*Rr+r), d]
__global__ void alphaw_kernel(const float* __restrict__ w, const float* __restrict__ X,
                              int Rr, int Dd, float* __restrict__ out, int ldc, int off) {
  int b = blockIdx.x;
  int d = blockIdx.y * 256 + threadIdx.x;
  if (d >= Dd) return;
  float s = 0;
  for (int r = 0; r < Rr; ++r) s += w[(long)b * Rr + r] * X[((long)(b * Rr + r)) * Dd + d];
  out[(long)b * ldc + off + d] = s;
}

// per batch row: lse[b] and argmax -> d_out[1+b]
__global__ void lse_argmax_kernel(const float* __restrict__ pred, float* __restrict__ lse,
                                  float* __restrict__ outv, int NA) {
  int b = blockIdx.x, tid = threadIdx.x;
  __shared__ float rv[256]; __shared__ int ri[256];
  float mx = -3.4e38f; int mi = 0x7fffffff;
  for (int j = tid; j < NA; j += 256) {
    float v = pred[(long)b * NA + j];
    if (v > mx) { mx = v; mi = j; }
  }
  rv[tid] = mx; ri[tid] = mi; __syncthreads();
  for (int o = 128; o > 0; o >>= 1) {
    if (tid < o) {
      if (rv[tid + o] > rv[tid] || (rv[tid + o] == rv[tid] && ri[tid + o] < ri[tid])) {
        rv[tid] = rv[tid + o]; ri[tid] = ri[tid + o];
      }
    }
    __syncthreads();
  }
  float gmx = rv[0]; int garg = ri[0]; __syncthreads();
  float s = 0;
  for (int j = tid; j < NA; j += 256) s += __expf(pred[(long)b * NA + j] - gmx);
  rv[tid] = s; __syncthreads();
  for (int o = 128; o > 0; o >>= 1) { if (tid < o) rv[tid] += rv[tid + o]; __syncthreads(); }
  if (tid == 0) { lse[b] = gmx + logf(rv[0]); outv[1 + b] = (float)garg; }
}

__global__ void loss_final_kernel(const float* __restrict__ pred, const float* __restrict__ lse,
                                  const int* __restrict__ labels, const float* __restrict__ scores,
                                  float* __restrict__ out, int NA, int Aa, int Bb) {
  int tid = threadIdx.x;
  __shared__ float rs[256], rc[256];
  float sum = 0, cnt = 0;
  for (int i = tid; i < Aa * Bb; i += 256) {
    int b = i % Bb;
    int lab = labels[i];
    float sc = scores[i];
    float nll = lse[b] - pred[(long)b * NA + lab];
    float w = sc * sc;
    float x = nll * w;
    if (sc >= 0.3f && x > 0.f) { sum += x; cnt += 1.f; }
  }
  rs[tid] = sum; rc[tid] = cnt; __syncthreads();
  for (int o = 128; o > 0; o >>= 1) {
    if (tid < o) { rs[tid] += rs[tid + o]; rc[tid] += rc[tid + o]; }
    __syncthreads();
  }
  if (tid == 0) out[0] = rs[0] / rc[0];
}

// ---------------------------------------------------------------------------
// Host orchestration
// ---------------------------------------------------------------------------
static void launch_gemm(hipStream_t s,
                        const float* A, int lda, const float* W,
                        const float* bias, const float* bias2,
                        const float* addend, int ldadd, const float* vec,
                        float* C, int ldc, int M, int N, int K, int mode) {
  dim3 grid((N + TN - 1) / TN, (M + TM - 1) / TM);
  gemm_wmma_kernel<<<grid, 256, 0, s>>>(A, lda, W, bias, bias2, addend, ldadd, vec,
                                        C, ldc, M, N, K, mode);
}
static void launch_fill(hipStream_t s, float* p, long n, float v, const float* addp) {
  fill_kernel<<<(unsigned)((n + 255) / 256), 256, 0, s>>>(p, n, v, addp);
}

extern "C" void kernel_launch(void* const* d_in, const int* in_sizes, int n_in,
                              void* d_out, int out_size, void* d_ws, size_t ws_size,
                              hipStream_t stream) {
  (void)in_sizes; (void)n_in; (void)out_size; (void)ws_size;
  const int Bv = 128, QL = 23, R = 196, E = 300, Hh = 512, Dd = 1024, RE = 2048, NA = 3129, Aa = 10;
  const int MQ = Bv * QL;   // 2944
  const int MR = Bv * R;    // 25088

  const float* images = (const float*)d_in[0];
  const int*   q_idxs = (const int*)d_in[1];
  const int*   labels = (const int*)d_in[2];
  const float* scores = (const float*)d_in[3];
#define PP(i) ((const float*)d_in[4 + (i)])
  const float *word_emb = PP(0);
  const float *Wih_f = PP(1),  *Whh_f = PP(2), *bih_f = PP(3), *bhh_f = PP(4);
  const float *Wih_b = PP(5),  *Whh_b = PP(6), *bih_b = PP(7), *bhh_b = PP(8);
  const float *Ienc_W = PP(9), *Ienc_b = PP(10);
  const float *Ie_W1 = PP(11), *Ie_b1 = PP(12), *Ie_W2 = PP(13), *Ie_b2 = PP(14);
  const float *Qe_W1 = PP(15), *Qe_b1 = PP(16), *Qe_W2 = PP(17), *Qe_b2 = PP(18);
  const float *QIL_W = PP(19), *QIL_b = PP(20), *QIR_W = PP(21), *QIR_b = PP(22);
  const float *IQL_W = PP(23), *IQL_b = PP(24), *IQR_W = PP(25), *IQR_b = PP(26);
  const float *QLs_W = PP(27), *QLs_b = PP(28), *QRs_W = PP(29), *QRs_b = PP(30);
  const float *ILs_W = PP(31), *ILs_b = PP(32), *IRs_W = PP(33), *IRs_b = PP(34);
  const float *WI_W = PP(35),  *WI_b = PP(36),  *WII_W = PP(37), *WII_b = PP(38);
  const float *WIQ_W = PP(39), *WIQ_b = PP(40);
  const float *WQ_W = PP(41),  *WQ_b = PP(42),  *WQQ_W = PP(43), *WQQ_b = PP(44);
  const float *WQI_W = PP(45), *WQI_b = PP(46);
  const float *D_W1 = PP(47),  *D_b1 = PP(48),  *D_W2 = PP(49),  *D_b2 = PP(50);
#undef PP

  float* wsf = (float*)d_ws;
  size_t off = 0;
  auto alloc = [&](size_t n) { float* p = wsf + off; off += n; return p; };
  float* enc_r  = alloc((size_t)MR * RE);
  float* bufR1  = alloc((size_t)MR * 512);
  float* bufR2  = alloc((size_t)MR * 512);
  float* xg_f   = alloc((size_t)MQ * 2048);
  float* xg_b   = alloc((size_t)MQ * 2048);
  float* enc_q  = alloc((size_t)MQ * 1024);
  float* q_emb  = alloc((size_t)MQ * E);
  float* bufQ1  = alloc((size_t)MQ * 512);
  float* bufQ2  = alloc((size_t)MQ * 512);
  float* gates_f= alloc((size_t)Bv * 2048);
  float* gates_b= alloc((size_t)Bv * 2048);
  float* c_f    = alloc((size_t)Bv * 512);
  float* c_b    = alloc((size_t)Bv * 512);
  float* zerosb = alloc((size_t)Bv * 512);
  float* normq  = alloc(MQ);
  float* normr  = alloc(MR);
  float* Sq     = alloc((size_t)Bv * 512);
  float* Sr     = alloc((size_t)Bv * 512);
  float* q_ent  = alloc(MQ);
  float* q_self = alloc(MQ);
  float* qi     = alloc(MQ);
  float* i_ent  = alloc(MR);
  float* i_self = alloc(MR);
  float* iq     = alloc(MR);
  float* bI     = alloc(MR);
  float* bQ     = alloc(MQ);
  float* cat    = alloc((size_t)Bv * 3072);
  float* hid    = alloc((size_t)Bv * NA);
  float* pred   = alloc((size_t)Bv * NA);
  float* lse    = alloc(Bv);
  float* outv   = (float*)d_out;

  // zero c_f, c_b, zerosb (contiguous)
  launch_fill(stream, c_f, (long)3 * Bv * 512, 0.f, nullptr);

  // 1) embedding gather
  embed_kernel<<<MQ, 128, 0, stream>>>(word_emb, q_idxs, q_emb, E);

  // 2) xg = q_emb @ Wih^T + (bih + bhh)
  launch_gemm(stream, q_emb, E, Wih_f, bih_f, bhh_f, nullptr, 0, nullptr, xg_f, 2048, MQ, 2048, E, 0);
  launch_gemm(stream, q_emb, E, Wih_b, bih_b, bhh_b, nullptr, 0, nullptr, xg_b, 2048, MQ, 2048, E, 0);

  // 3) LSTM scans (forward t=0..22, backward t=22..0); h written into enc_q halves
  for (int t = 0; t < QL; ++t) {
    const float* hprev_f = (t == 0) ? zerosb : (enc_q + (size_t)(t - 1) * 1024);
    int lda_f = (t == 0) ? 512 : QL * 1024;
    launch_gemm(stream, hprev_f, lda_f, Whh_f, nullptr, nullptr,
                xg_f + (size_t)t * 2048, QL * 2048, nullptr, gates_f, 2048, Bv, 2048, Hh, 0);
    lstm_pointwise_kernel<<<(Bv * 512 + 255) / 256, 256, 0, stream>>>(gates_f, c_f, enc_q, t, 0);

    int tb = QL - 1 - t;
    const float* hprev_b = (t == 0) ? zerosb : (enc_q + (size_t)(tb + 1) * 1024 + 512);
    int lda_b = (t == 0) ? 512 : QL * 1024;
    launch_gemm(stream, hprev_b, lda_b, Whh_b, nullptr, nullptr,
                xg_b + (size_t)tb * 2048, QL * 2048, nullptr, gates_b, 2048, Bv, 2048, Hh, 0);
    lstm_pointwise_kernel<<<(Bv * 512 + 255) / 256, 256, 0, stream>>>(gates_b, c_b, enc_q, tb, 512);
  }

  // 4) q_ent (fused entropy MLP: reduce mode)
  launch_fill(stream, q_ent, MQ, 0.f, Qe_b2);
  launch_gemm(stream, enc_q, Dd, Qe_W1, Qe_b1, nullptr, nullptr, 0, Qe_W2, q_ent, 0, MQ, Dd, Dd, 2);

  // 5) q_self
  launch_gemm(stream, enc_q, Dd, QLs_W, QLs_b, nullptr, nullptr, 0, nullptr, bufQ1, 512, MQ, 512, Dd, 0);
  launch_gemm(stream, enc_q, Dd, QRs_W, QRs_b, nullptr, nullptr, 0, nullptr, bufQ2, 512, MQ, 512, Dd, 0);
  rowdot_nrm_kernel<<<MQ, 256, 0, stream>>>(bufQ1, bufQ2, q_self, (float)QL);

  // 6) q-side cross projections: bufQ1 = QIL(enc_q), bufQ2 = IQR(enc_q); Sq = sum_q nrm(IQR)
  launch_gemm(stream, enc_q, Dd, QIL_W, QIL_b, nullptr, nullptr, 0, nullptr, bufQ1, 512, MQ, 512, Dd, 0);
  launch_gemm(stream, enc_q, Dd, IQR_W, IQR_b, nullptr, nullptr, 0, nullptr, bufQ2, 512, MQ, 512, Dd, 0);
  row_norms_kernel<<<MQ, 256, 0, stream>>>(bufQ2, normq);
  sum_nrm_kernel<<<dim3(Bv, 2), 256, 0, stream>>>(bufQ2, normq, Sq, QL);

  // 7) enc_r = leaky(images @ Ienc_W^T + b)
  launch_gemm(stream, images, RE, Ienc_W, Ienc_b, nullptr, nullptr, 0, nullptr, enc_r, RE, MR, RE, RE, 1);

  // 8) i_ent (fused entropy MLP)
  launch_fill(stream, i_ent, MR, 0.f, Ie_b2);
  launch_gemm(stream, enc_r, RE, Ie_W1, Ie_b1, nullptr, nullptr, 0, Ie_W2, i_ent, 0, MR, RE, RE, 2);

  // 9) i_self
  launch_gemm(stream, enc_r, RE, ILs_W, ILs_b, nullptr, nullptr, 0, nullptr, bufR1, 512, MR, 512, RE, 0);
  launch_gemm(stream, enc_r, RE, IRs_W, IRs_b, nullptr, nullptr, 0, nullptr, bufR2, 512, MR, 512, RE, 0);
  rowdot_nrm_kernel<<<MR, 256, 0, stream>>>(bufR1, bufR2, i_self, (float)R);

  // 10) iq = dot(nrm(IQL(enc_r)), Sq[b])
  launch_gemm(stream, enc_r, RE, IQL_W, IQL_b, nullptr, nullptr, 0, nullptr, bufR1, 512, MR, 512, RE, 0);
  rowdot_vec_kernel<<<MR, 256, 0, stream>>>(bufR1, Sq, iq, R);

  // 11) qi = dot(nrm(QIL(enc_q)), Sr[b]); Sr from QIR(enc_r)
  launch_gemm(stream, enc_r, RE, QIR_W, QIR_b, nullptr, nullptr, 0, nullptr, bufR2, 512, MR, 512, RE, 0);
  row_norms_kernel<<<MR, 256, 0, stream>>>(bufR2, normr);
  sum_nrm_kernel<<<dim3(Bv, 2), 256, 0, stream>>>(bufR2, normr, Sr, R);
  rowdot_vec_kernel<<<MQ, 256, 0, stream>>>(bufQ1, Sr, qi, QL);

  // 12) b_I = softmax(i_ent@WI^T + iq@WIQ^T + i_self@WII^T + biases)
  launch_gemm(stream, i_ent,  R, WI_W,  WI_b,  nullptr, nullptr, 0, nullptr, bI, R, Bv, R, R, 0);
  launch_gemm(stream, iq,     R, WIQ_W, WIQ_b, nullptr, bI, R, nullptr, bI, R, Bv, R, R, 0);
  launch_gemm(stream, i_self, R, WII_W, WII_b, nullptr, bI, R, nullptr, bI, R, Bv, R, R, 0);
  softmax_rows_kernel<<<Bv, 256, 0, stream>>>(bI, R);

  // 13) b_Q
  launch_gemm(stream, q_ent,  QL, WQ_W,  WQ_b,  nullptr, nullptr, 0, nullptr, bQ, QL, Bv, QL, QL, 0);
  launch_gemm(stream, qi,     QL, WQI_W, WQI_b, nullptr, bQ, QL, nullptr, bQ, QL, Bv, QL, QL, 0);
  launch_gemm(stream, q_self, QL, WQQ_W, WQQ_b, nullptr, bQ, QL, nullptr, bQ, QL, Bv, QL, QL, 0);
  softmax_rows_kernel<<<Bv, 256, 0, stream>>>(bQ, QL);

  // 14) attention pooling -> cat(128 x 3072)
  alphaw_kernel<<<dim3(Bv, RE / 256), 256, 0, stream>>>(bI, enc_r, R, RE, cat, 3072, 0);
  alphaw_kernel<<<dim3(Bv, Dd / 256), 256, 0, stream>>>(bQ, enc_q, QL, Dd, cat, 3072, RE);

  // 15) classifier head
  launch_gemm(stream, cat, 3072, D_W1, D_b1, nullptr, nullptr, 0, nullptr, hid, NA, Bv, NA, 3072, 1);
  launch_gemm(stream, hid, NA,   D_W2, D_b2, nullptr, nullptr, 0, nullptr, pred, NA, Bv, NA, NA, 0);

  // 16) log-softmax stats + argmax, then weighted NLL loss
  lse_argmax_kernel<<<Bv, 256, 0, stream>>>(pred, lse, outv, NA);
  loss_final_kernel<<<1, 256, 0, stream>>>(pred, lse, labels, scores, outv, NA, Aa, Bv);
}
